// SimOTA_44736379355416
// MI455X (gfx1250) — matrix operand order repaired and checked by reading, not compiled, not run on hardware
//
#include <hip/hip_runtime.h>
#include <cstdint>
#include <cstddef>

// ---------------- problem constants ----------------
#define NUM_B 16
#define NUM_A 8400
#define NUM_T 128
#define NUM_C 80
#define TOPK 10
#define PRED_STRIDE 85
#define REC_WORDS 56          // 224 bytes / 4: box(4f) + l1sum(1f) + pad(3f) + u[96] f16 (48f)
#define REC_BYTES 224
#define U_OFF_BYTES 32
#define CHUNK 64
#define NCHUNK ((NUM_A + CHUNK - 1) / CHUNK)   // 132

// ---------------- vector types ----------------
typedef _Float16 v16h __attribute__((ext_vector_type(16)));
typedef float    v8f  __attribute__((ext_vector_type(8)));
typedef unsigned int u32x4 __attribute__((ext_vector_type(4)));
typedef int      i32x4 __attribute__((ext_vector_type(4)));
typedef int      i32x8 __attribute__((ext_vector_type(8)));

#if __has_builtin(__builtin_amdgcn_wmma_f32_16x16x32_f16)
#define USE_WMMA 1
#else
#define USE_WMMA 0
#endif

#if __has_builtin(__builtin_amdgcn_tensor_load_to_lds) && __has_builtin(__builtin_amdgcn_s_wait_tensorcnt)
#define USE_TDM 1
#else
#define USE_TDM 0
#endif

// ---------------- shared math helpers (must be bit-identical across kernels) ----------------
__device__ __forceinline__ float pair_iou(float tx0, float ty0, float tx1, float ty1,
                                          float px0, float py0, float px1, float py1) {
  float lx = fmaxf(tx0, px0), ly = fmaxf(ty0, py0);
  float rx = fminf(tx1, px1), ry = fminf(ty1, py1);
  float w = fmaxf(rx - lx, 0.f), h = fmaxf(ry - ly, 0.f);
  float inter = w * h;
  float at = fmaxf(tx1 - tx0, 0.f) * fmaxf(ty1 - ty0, 0.f);
  float ap = fmaxf(px1 - px0, 0.f) * fmaxf(py1 - py0, 0.f);
  return inter / (at + ap - inter + 1e-8f);
}

__device__ __forceinline__ float pair_cost(float uval, float l1sum, float iou, int ibc, int ib) {
  float iou_cost = -logf(iou + 1e-8f);
  float c = -(l1sum + uval) + 3.0f * iou_cost;
  c += ibc ? 0.f : 1.0e5f;
  c += ib  ? 0.f : 1.0e9f;
  return c;
}

__device__ __forceinline__ float bce_(float x, float t) {
  return fmaxf(x, 0.f) - x * t + log1pf(expf(-fabsf(x)));
}

__device__ __forceinline__ float ciou_(float px0, float py0, float px1, float py1,
                                       float tx0, float ty0, float tx1, float ty1) {
  float lx = fmaxf(px0, tx0), ly = fmaxf(py0, ty0);
  float rx = fminf(px1, tx1), ry = fminf(py1, ty1);
  float w = fmaxf(rx - lx, 0.f), h = fmaxf(ry - ly, 0.f);
  float inter = w * h;
  float wp = px1 - px0, hp = py1 - py0, wt = tx1 - tx0, ht = ty1 - ty0;
  float iou = inter / (wp * hp + wt * ht - inter + 1e-8f);
  float cw = fmaxf(px1, tx1) - fminf(px0, tx0);
  float chh = fmaxf(py1, ty1) - fminf(py0, ty0);
  float c2 = cw * cw + chh * chh + 1e-8f;
  float dx = (px0 + px1) - (tx0 + tx1), dy = (py0 + py1) - (ty0 + ty1);
  float rho2 = (dx * dx + dy * dy) * 0.25f;
  const float PI = 3.14159265358979323846f;
  float ang = atanf(wt / (ht + 1e-8f)) - atanf(wp / (hp + 1e-8f));
  float v = (4.0f / (PI * PI)) * ang * ang;
  float alpha = v / (1.0f - iou + v + 1e-8f);
  return iou - rho2 / c2 - alpha * v;
}

// ---------------- K0: init match state ----------------
__global__ void k0_init(uint32_t* __restrict__ cnt, uint32_t* __restrict__ mt, int n) {
  int i = blockIdx.x * blockDim.x + threadIdx.x;
  if (i < n) { cnt[i] = 0u; mt[i] = 0xFFFFFFFFu; }
}

// ---------------- K1: per-anchor precompute ----------------
__global__ __launch_bounds__(256) void k1_prep(
    const float* __restrict__ pred, const float* __restrict__ target,
    const float* __restrict__ grid, const float* __restrict__ stridem,
    float* __restrict__ rec, uint32_t* __restrict__ inb_bits,
    uint32_t* __restrict__ inc_bits, uint32_t* __restrict__ in_box) {
  __shared__ float tg[NUM_T * 6];
  int b = blockIdx.x;
  int a = blockIdx.y * blockDim.x + threadIdx.x;
  for (int i = threadIdx.x; i < NUM_T * 6; i += blockDim.x)
    tg[i] = target[(size_t)b * NUM_T * 6 + i];
  __syncthreads();
  if (a >= NUM_A) return;
  size_t ai = (size_t)b * NUM_A + a;
  const float* p = pred + ai * PRED_STRIDE;
  __builtin_prefetch(p + PRED_STRIDE, 0, 1);   // global_prefetch_b8 of next anchor row
  float px0 = p[0], py0 = p[1], px1 = p[2], py1 = p[3];
  float sobj = 1.f / (1.f + expf(-p[4]));
  float l1sum = 0.f;
  _Float16 uh[96];
  for (int c = 0; c < 96; c++) {
    float u = 0.f;
    if (c < NUM_C) {
      float l = p[5 + c];
      float sc = 1.f / (1.f + expf(-l));
      float prob = sqrtf(sobj * sc);
      float lp = fmaxf(logf(prob), -100.f);
      float l1 = fmaxf(log1pf(-prob), -100.f);
      l1sum += l1;
      u = lp - l1;
    }
    uh[c] = (_Float16)u;
  }
  float gx = grid[a * 2 + 0], gy = grid[a * 2 + 1], s = stridem[a];
  float xc = (gx + 0.5f) * s, yc = (gy + 0.5f) * s;
  uint32_t wb[4] = {0, 0, 0, 0}, wc[4] = {0, 0, 0, 0};
  for (int t = 0; t < NUM_T; t++) {
    const float* tr = &tg[t * 6];
    float tx0 = tr[2], ty0 = tr[3], tx1 = tr[4], ty1 = tr[5];
    float dmin = fminf(fminf(xc - tx0, yc - ty0), fminf(tx1 - xc, ty1 - yc));
    float txc = (tx0 + tx1) * 0.5f, tyc = (ty0 + ty1) * 0.5f;
    int inb = dmin > 0.f;
    int inc = fmaxf(fabsf(xc - txc), fabsf(yc - tyc)) < 2.5f * s;
    if (inb) wb[t >> 5] |= (1u << (t & 31));
    if (inc) wc[t >> 5] |= (1u << (t & 31));
  }
  uint32_t anyb = wb[0] | wb[1] | wb[2] | wb[3];
  uint32_t anyc = wc[0] | wc[1] | wc[2] | wc[3];
  in_box[ai] = (anyb | anyc) ? 1u : 0u;
  for (int i = 0; i < 4; i++) { inb_bits[ai * 4 + i] = wb[i]; inc_bits[ai * 4 + i] = wc[i]; }
  float* r = rec + ai * REC_WORDS;
  r[0] = px0; r[1] = py0; r[2] = px1; r[3] = py1;
  r[4] = l1sum; r[5] = 0.f; r[6] = 0.f; r[7] = 0.f;
  _Float16* ur = (_Float16*)(r + 8);
  for (int c = 0; c < 96; c++) ur[c] = uh[c];
}

// ---------------- K2: cost + top-k + matching scatter ----------------
__global__ __launch_bounds__(128) void k2_cost(
    const float* __restrict__ rec, const float* __restrict__ target,
    const uint32_t* __restrict__ inb_bits, const uint32_t* __restrict__ inc_bits,
    const uint32_t* __restrict__ in_box,
    float* __restrict__ row_max_iou, uint32_t* __restrict__ match_cnt,
    uint32_t* __restrict__ match_t) {
  __shared__ float srec[2][CHUNK * REC_WORDS];   // 28672 B double-buffered anchor records
  __shared__ float cost_s[CHUNK * 16];
  __shared__ float iou_s[CHUNK * 16];
  __shared__ float tbox_s[16][4];
  __shared__ int   cid_s[16];
  __shared__ float mv_cost[128 * TOPK];
  __shared__ int   mi_cost[128 * TOPK];
  __shared__ float mv_iou[128 * TOPK];
  __shared__ float mmax_iou[128];

  int b = blockIdx.x;
  int tt = blockIdx.y;              // target tile (16 targets)
  int tid = threadIdx.x;
  int lane = tid & 31;
  int wv = tid >> 5;                // wave id 0..3
  int mrow = lane & 15;
  int hsel = lane >> 4;             // 0/1 lane half

  // make the staging buffer "written" from the compiler's POV + real zero-fill
  for (int i = tid; i < 2 * CHUNK * REC_WORDS; i += 128) srec[0][i] = 0.f;
  if (tid < 16) {
    const float* tr = target + ((size_t)b * NUM_T + tt * 16 + tid) * 6;
    tbox_s[tid][0] = tr[2]; tbox_s[tid][1] = tr[3];
    tbox_s[tid][2] = tr[4]; tbox_s[tid][3] = tr[5];
    cid_s[tid] = (int)tr[1];
  }
  __syncthreads();

#if USE_WMMA
  // One-hot A fragments: A[m, k] = (class(k) == cid[m]); contiguous k = 16*hsel + h, matching the B loads.
  union { v16h v; _Float16 e[16]; } af[3];
  int mycid = cid_s[mrow];
  for (int s = 0; s < 3; s++)
    for (int h = 0; h < 16; h++) {
      int K = hsel * 16 + h;
      af[s].e[h] = ((s * 32 + K) == mycid) ? (_Float16)1.0f : (_Float16)0.0f;
    }
#endif

  float tkc_v[TOPK]; int tkc_i[TOPK]; float tki_v[TOPK]; float imax = 0.f;
  for (int k = 0; k < TOPK; k++) { tkc_v[k] = 3.4e38f; tkc_i[k] = 0x7fffffff; tki_v[k] = -1.f; }

#if USE_TDM
  auto issue_load = [&](int ch) {
    int buf = ch & 1;
    // low 32 bits of a flat shared pointer == workgroup-relative LDS byte offset
    uint32_t lds_addr = (uint32_t)(uintptr_t)(&srec[buf][0]);
    uint64_t ga = (uint64_t)(uintptr_t)rec +
                  ((uint64_t)((size_t)b * NUM_A + (size_t)ch * CHUNK)) * REC_BYTES;
    int rema = NUM_A - ch * CHUNK; if (rema > CHUNK) rema = CHUNK;
    uint32_t td0 = (uint32_t)rema * REC_WORDS;        // valid dwords (OOB -> zero fill)
    u32x4 g0 = { 1u /*count=1*/, lds_addr, (uint32_t)ga,
                 (uint32_t)((ga >> 32) & 0x1FFFFFFu) | (2u << 30) /*type=image*/ };
    i32x8 g1;
    g1[0] = 0x00020000;                               // data_size = 4B, no multicast
    g1[1] = (int)((td0 & 0xFFFFu) << 16);             // tensor_dim0 lo16 @bit48
    g1[2] = (int)((td0 >> 16) | (1u << 16));          // tensor_dim0 hi16 | tensor_dim1=1
    g1[3] = (int)((uint32_t)(CHUNK * REC_WORDS) << 16); // tile_dim0 = 3584
    g1[4] = 0;
    g1[5] = (int)td0;                                 // tensor_dim0_stride lo32
    g1[6] = 0; g1[7] = 0;
    i32x4 z4 = {0, 0, 0, 0};
#if defined(__clang_major__) && (__clang_major__ >= 23)
    i32x8 z8 = {0, 0, 0, 0, 0, 0, 0, 0};
    __builtin_amdgcn_tensor_load_to_lds(g0, g1, z4, z4, z8, 0);
#else
    __builtin_amdgcn_tensor_load_to_lds(g0, g1, z4, z4, 0);
#endif
  };
  if (wv == 0) issue_load(0);
  asm volatile("" ::: "memory");
#endif

  for (int ch = 0; ch < NCHUNK; ch++) {
    int buf = ch & 1;
#if USE_TDM
    if (wv == 0) {
      if (ch + 1 < NCHUNK) { issue_load(ch + 1); __builtin_amdgcn_s_wait_tensorcnt(1); }
      else                 { __builtin_amdgcn_s_wait_tensorcnt(0); }
    }
    asm volatile("" ::: "memory");
    __syncthreads();
#else
    {
      int base = ch * CHUNK;
      int rema = NUM_A - base; if (rema > CHUNK) rema = CHUNK;
      int words = rema * REC_WORDS;
      const float* src = rec + ((size_t)b * NUM_A + base) * REC_WORDS;
      for (int i = tid; i < CHUNK * REC_WORDS; i += 128)
        srec[buf][i] = (i < words) ? src[i] : 0.f;
    }
    __syncthreads();
#endif
    // --- each wave computes one 16(anchor) x 16(target) tile ---
    int al = wv * 16 + mrow;                 // local anchor index in chunk
    int aglob = ch * CHUNK + al;
    bool valid = (aglob < NUM_A);
    const float* rbase = &srec[buf][al * REC_WORDS];

    union { v8f v; float e[8]; } cu;
#if USE_WMMA
    v8f acc = {0.f, 0.f, 0.f, 0.f, 0.f, 0.f, 0.f, 0.f};
#pragma unroll
    for (int s = 0; s < 3; s++) {
      const uint32_t* up =
          (const uint32_t*)((const char*)rbase + U_OFF_BYTES + (s * 32 + hsel * 16) * 2);
      union { v16h v; uint32_t w[8]; } bu;
#pragma unroll
      for (int i = 0; i < 8; i++) bu.w[i] = up[i];
      acc = __builtin_amdgcn_wmma_f32_16x16x32_f16(false, af[s].v, false, bu.v,
                                                   (short)0, acc, false, false);
    }
    cu.v = acc;   // cu.e[r] = u[anchor, cid[target r + 8*hsel]]
#else
    {
      const _Float16* ur = (const _Float16*)((const char*)rbase + U_OFF_BYTES);
#pragma unroll
      for (int r = 0; r < 8; r++) cu.e[r] = (float)ur[cid_s[r + 8 * hsel]];
    }
#endif
    float px0 = rbase[0], py0 = rbase[1], px1 = rbase[2], py1 = rbase[3];
    float l1sum = rbase[4];
    size_t ai = (size_t)b * NUM_A + (size_t)(valid ? aglob : 0);
    uint32_t inbw = inb_bits[ai * 4 + (tt >> 1)];
    uint32_t incw = inc_bits[ai * 4 + (tt >> 1)];
    uint32_t ib = in_box[ai];
    int bitbase = (tt & 1) * 16;
#pragma unroll
    for (int r = 0; r < 8; r++) {
      int tl = r + 8 * hsel;
      float iou = 0.f;
      if (ib) iou = pair_iou(tbox_s[tl][0], tbox_s[tl][1], tbox_s[tl][2], tbox_s[tl][3],
                             px0, py0, px1, py1);
      int ibc = (int)(((inbw & incw) >> (bitbase + tl)) & 1u);
      float c = pair_cost(cu.e[r], l1sum, iou, ibc, (int)ib);
      if (!valid) { c = 3.0e38f; iou = 0.f; }
      cost_s[al * 16 + tl] = c;
      iou_s[al * 16 + tl] = iou;
    }
    __syncthreads();

    // --- streaming top-k: 8 threads per target, interleaved anchors ---
    int mtgt = tid >> 3, sub = tid & 7;
#pragma unroll
    for (int q = 0; q < 8; q++) {
      int alx = sub + q * 8;
      int aidx = ch * CHUNK + alx;
      float c = cost_s[alx * 16 + mtgt];
      float io = iou_s[alx * 16 + mtgt];
      imax = fmaxf(imax, io);
      if (c < tkc_v[TOPK - 1] || (c == tkc_v[TOPK - 1] && aidx < tkc_i[TOPK - 1])) {
        int k = TOPK - 1;
        while (k > 0 && (c < tkc_v[k - 1] || (c == tkc_v[k - 1] && aidx < tkc_i[k - 1]))) {
          tkc_v[k] = tkc_v[k - 1]; tkc_i[k] = tkc_i[k - 1]; k--;
        }
        tkc_v[k] = c; tkc_i[k] = aidx;
      }
      if (io > tki_v[TOPK - 1]) {
        int k = TOPK - 1;
        while (k > 0 && io > tki_v[k - 1]) { tki_v[k] = tki_v[k - 1]; k--; }
        tki_v[k] = io;
      }
    }
    __syncthreads();
  }

  for (int k = 0; k < TOPK; k++) {
    mv_cost[tid * TOPK + k] = tkc_v[k];
    mi_cost[tid * TOPK + k] = tkc_i[k];
    mv_iou[tid * TOPK + k] = tki_v[k];
  }
  mmax_iou[tid] = imax;
  __syncthreads();

  if (tid < 16) {
    int mtgt = tid;
    int t_global = tt * 16 + mtgt;
    int cur[8]; for (int s = 0; s < 8; s++) cur[s] = 0;
    int fi[TOPK];
    for (int k = 0; k < TOPK; k++) {                 // 8-way merge (cost, ties -> low idx)
      int bs = -1; float bv = 0.f; int bi = 0;
      for (int s = 0; s < 8; s++) {
        int th = mtgt * 8 + s;
        if (cur[s] < TOPK) {
          float v = mv_cost[th * TOPK + cur[s]];
          int ii = mi_cost[th * TOPK + cur[s]];
          if (bs < 0 || v < bv || (v == bv && ii < bi)) { bs = s; bv = v; bi = ii; }
        }
      }
      fi[k] = bi; cur[bs]++;
    }
    for (int s = 0; s < 8; s++) cur[s] = 0;
    float isum = 0.f;
    for (int k = 0; k < TOPK; k++) {                 // 8-way merge (iou, largest)
      int bs = 0; float bv = -1.f;
      for (int s = 0; s < 8; s++) {
        int th = mtgt * 8 + s;
        if (cur[s] < TOPK) {
          float v = mv_iou[th * TOPK + cur[s]];
          if (v > bv) { bv = v; bs = s; }
        }
      }
      isum += bv; cur[bs]++;
    }
    float rmax = 0.f;
    for (int s = 0; s < 8; s++) rmax = fmaxf(rmax, mmax_iou[mtgt * 8 + s]);
    row_max_iou[(size_t)b * NUM_T + t_global] = rmax;
    int dk = (int)isum; if (dk < 1) dk = 1; if (dk > TOPK) dk = TOPK;
    for (int k = 0; k < dk; k++) {
      size_t ai = (size_t)b * NUM_A + fi[k];
      if (in_box[ai]) {
        atomicAdd(&match_cnt[ai], 1u);
        atomicMin(&match_t[ai], (uint32_t)t_global);
      }
    }
  }
}

// ---------------- K3: matching finalize + loss terms (deterministic block partials) ---------
__global__ __launch_bounds__(256) void k3_loss(
    const float* __restrict__ pred, const float* __restrict__ target,
    const float* __restrict__ rec,
    const uint32_t* __restrict__ inb_bits, const uint32_t* __restrict__ inc_bits,
    const uint32_t* __restrict__ in_box, const float* __restrict__ row_max_iou,
    const uint32_t* __restrict__ match_cnt, const uint32_t* __restrict__ match_t,
    float* __restrict__ partials) {
  __shared__ float tb[NUM_T][4];
  __shared__ int   tc[NUM_T];
  __shared__ float rm[NUM_T];
  __shared__ float r0[256], r1[256], r2[256], r3[256];
  int b = blockIdx.x;
  int tid = threadIdx.x;
  for (int t = tid; t < NUM_T; t += 256) {
    const float* tr = target + ((size_t)b * NUM_T + t) * 6;
    tb[t][0] = tr[2]; tb[t][1] = tr[3]; tb[t][2] = tr[4]; tb[t][3] = tr[5];
    tc[t] = (int)tr[1];
    rm[t] = row_max_iou[(size_t)b * NUM_T + t];
  }
  __syncthreads();
  int a = blockIdx.y * 256 + tid;
  float s_m = 0.f, s_lbox = 0.f, s_lobj = 0.f, s_lcls = 0.f;
  if (a < NUM_A) {
    size_t ai = (size_t)b * NUM_A + a;
    const float* p = pred + ai * PRED_STRIDE;
    float px0 = p[0], py0 = p[1], px1 = p[2], py1 = p[3], obj = p[4];
    uint32_t ib = in_box[ai];
    float objt = 0.f;
    if (ib) {
      float pm = 0.f;
      for (int t = 0; t < NUM_T; t++) {
        float iou = pair_iou(tb[t][0], tb[t][1], tb[t][2], tb[t][3], px0, py0, px1, py1);
        pm = fmaxf(pm, iou / (rm[t] + 1e-8f));
      }
      objt = fminf(fmaxf(pm, 0.f), 1.f);
    }
    s_lobj = bce_(obj, objt);
    uint32_t cnt = match_cnt[ai];
    if (cnt > 0u) {
      int tp;
      if (cnt == 1u) tp = (int)match_t[ai];
      else {
        float l1sum = rec[ai * REC_WORDS + 4];
        const _Float16* ur = (const _Float16*)(rec + ai * REC_WORDS + 8);
        uint32_t wbv[4], wcv[4];
        for (int i = 0; i < 4; i++) { wbv[i] = inb_bits[ai * 4 + i]; wcv[i] = inc_bits[ai * 4 + i]; }
        float best = 3.4e38f; int bi = 0;
        for (int t = 0; t < NUM_T; t++) {
          float iou = ib ? pair_iou(tb[t][0], tb[t][1], tb[t][2], tb[t][3], px0, py0, px1, py1) : 0.f;
          int ibc = (int)(((wbv[t >> 5] & wcv[t >> 5]) >> (t & 31)) & 1u);
          float c = pair_cost((float)ur[tc[t]], l1sum, iou, ibc, (int)ib);
          if (c < best) { best = c; bi = t; }
        }
        tp = bi;
      }
      s_m = 1.f;
      s_lbox = 1.f - ciou_(px0, py0, px1, py1, tb[tp][0], tb[tp][1], tb[tp][2], tb[tp][3]);
      int cidt = tc[tp];
      float cl = 0.f;
      for (int c = 0; c < NUM_C; c++) cl += bce_(p[5 + c], (c == cidt) ? 1.f : 0.f);
      s_lcls = cl * (1.0f / NUM_C);
    }
  }
  r0[tid] = s_m; r1[tid] = s_lbox; r2[tid] = s_lobj; r3[tid] = s_lcls;
  __syncthreads();
  for (int off = 128; off > 0; off >>= 1) {
    if (tid < off) {
      r0[tid] += r0[tid + off]; r1[tid] += r1[tid + off];
      r2[tid] += r2[tid + off]; r3[tid] += r3[tid + off];
    }
    __syncthreads();
  }
  if (tid == 0) {
    int pbi = blockIdx.x * gridDim.y + blockIdx.y;
    partials[pbi * 4 + 0] = r0[0]; partials[pbi * 4 + 1] = r1[0];
    partials[pbi * 4 + 2] = r2[0]; partials[pbi * 4 + 3] = r3[0];
  }
}

// ---------------- K4: final deterministic reduction ----------------
__global__ __launch_bounds__(256) void k4_final(const float* __restrict__ partials, int nblk,
                                                float* __restrict__ out) {
  __shared__ float s0[256], s1[256], s2[256], s3[256];
  int tid = threadIdx.x;
  float a0 = 0.f, a1 = 0.f, a2 = 0.f, a3 = 0.f;
  for (int i = tid; i < nblk; i += 256) {
    a0 += partials[i * 4 + 0]; a1 += partials[i * 4 + 1];
    a2 += partials[i * 4 + 2]; a3 += partials[i * 4 + 3];
  }
  s0[tid] = a0; s1[tid] = a1; s2[tid] = a2; s3[tid] = a3;
  __syncthreads();
  for (int off = 128; off > 0; off >>= 1) {
    if (tid < off) {
      s0[tid] += s0[tid + off]; s1[tid] += s1[tid + off];
      s2[tid] += s2[tid + off]; s3[tid] += s3[tid + off];
    }
    __syncthreads();
  }
  if (tid == 0) {
    float n = fmaxf(s0[0], 1.0f);
    float lbox = 0.05f * s1[0] / n;
    float lobj = s2[0] / (float)(NUM_B * NUM_A);
    float lcls = 0.5f * s3[0] / n;
    out[0] = lbox + lobj + lcls; out[1] = lbox; out[2] = lobj; out[3] = lcls;
  }
}

// ---------------- host launcher ----------------
extern "C" void kernel_launch(void* const* d_in, const int* in_sizes, int n_in,
                              void* d_out, int out_size, void* d_ws, size_t ws_size,
                              hipStream_t stream) {
  (void)in_sizes; (void)n_in; (void)out_size; (void)ws_size;
  const float* pred = (const float*)d_in[0];
  const float* target = (const float*)d_in[1];
  const float* grid = (const float*)d_in[2];
  const float* stridem = (const float*)d_in[3];

  char* ws = (char*)d_ws;
  size_t off = 0;
  auto alloc = [&](size_t bytes) -> void* {
    void* p = ws + off;
    off = (off + bytes + 255) & ~(size_t)255;
    return p;
  };
  float*    rec    = (float*)alloc((size_t)NUM_B * NUM_A * REC_BYTES);   // ~30.1 MB
  uint32_t* inb    = (uint32_t*)alloc((size_t)NUM_B * NUM_A * 16);
  uint32_t* incb   = (uint32_t*)alloc((size_t)NUM_B * NUM_A * 16);
  uint32_t* inbox  = (uint32_t*)alloc((size_t)NUM_B * NUM_A * 4);
  float*    rowmax = (float*)alloc((size_t)NUM_B * NUM_T * 4);
  uint32_t* mcnt   = (uint32_t*)alloc((size_t)NUM_B * NUM_A * 4);
  uint32_t* mt     = (uint32_t*)alloc((size_t)NUM_B * NUM_A * 4);
  const int ABLK = (NUM_A + 255) / 256;           // 33
  const int NB3 = NUM_B * ABLK;                   // 528
  float* partials = (float*)alloc((size_t)NB3 * 4 * 4);

  k0_init<<<(NUM_B * NUM_A + 255) / 256, 256, 0, stream>>>(mcnt, mt, NUM_B * NUM_A);
  k1_prep<<<dim3(NUM_B, ABLK), 256, 0, stream>>>(pred, target, grid, stridem,
                                                 rec, inb, incb, inbox);
  k2_cost<<<dim3(NUM_B, NUM_T / 16), 128, 0, stream>>>(rec, target, inb, incb, inbox,
                                                       rowmax, mcnt, mt);
  k3_loss<<<dim3(NUM_B, ABLK), 256, 0, stream>>>(pred, target, rec, inb, incb, inbox,
                                                 rowmax, mcnt, mt, partials);
  k4_final<<<1, 256, 0, stream>>>(partials, NB3, (float*)d_out);
}
